// LocationAwareAttention_17497696764361
// MI455X (gfx1250) — matrix operand order repaired and checked by reading, not compiled, not run on hardware
//
#include <hip/hip_runtime.h>
#include <hip/hip_bf16.h>
#include <cstdint>

// Problem constants
#define BB   32
#define VL   2048
#define HD   1024
#define DM   1024
#define CC   32

typedef __attribute__((ext_vector_type(16))) __bf16 v16bf;
typedef __attribute__((ext_vector_type(8)))  float  v8f;

union FragU { uint4 u[2]; v16bf v; };

__device__ __forceinline__ unsigned short f2bf(float x) {
    unsigned int u = __float_as_uint(x);
    unsigned int r = u + 0x7FFFu + ((u >> 16) & 1u);   // round-to-nearest-even
    return (unsigned short)(r >> 16);
}

// ---------------------------------------------------------------------------
// qpb[b][d] = bias[d] + sum_h query[b,0,h] * w_q[d,h]
// ---------------------------------------------------------------------------
__global__ __launch_bounds__(256) void qpb_kernel(
    const float* __restrict__ query, const float* __restrict__ w_q,
    const float* __restrict__ bias, float* __restrict__ qpb) {
    __shared__ float q[HD];
    const int b = blockIdx.x, tid = threadIdx.x;
    for (int i = tid; i < HD; i += 256) q[i] = query[b * HD + i];
    __syncthreads();
    for (int d = tid; d < DM; d += 256) {
        const float* wr = w_q + (size_t)d * HD;
        float acc = bias[d];
        #pragma unroll 4
        for (int k = 0; k < HD; ++k) acc += q[k] * wr[k];
        qpb[b * DM + d] = acc;
    }
}

// ---------------------------------------------------------------------------
// Pack w_v (DM x HD, f32) into bf16 B-fragment layout:
//   wvp[((nt*32 + kc)*32 + lane)*16 + j] = bf16(w_v[d, kc*32 + (lane>=16)*8 + j + (j>=8)*8])
//   d = nt*16 + (lane & 15)
// ---------------------------------------------------------------------------
__global__ __launch_bounds__(256) void pack_wv_kernel(
    const float* __restrict__ w_v, unsigned short* __restrict__ wvp) {
    const int i = blockIdx.x * 256 + threadIdx.x;      // 64*32*32 = 65536
    const int lane = i & 31, kc = (i >> 5) & 31, nt = i >> 10;
    const int d = nt * 16 + (lane & 15);
    const int kb = kc * 32 + ((lane >> 4) << 3);
    unsigned short* o = wvp + (size_t)i * 16;
    #pragma unroll
    for (int j = 0; j < 16; ++j) o[j] = f2bf(w_v[(size_t)d * HD + kb + j + ((j >> 3) << 3)]);
}

// ---------------------------------------------------------------------------
// Pack w_loc (DM x CC) into one K=32 B-fragment per n-tile.
// ---------------------------------------------------------------------------
__global__ __launch_bounds__(256) void pack_wloc_kernel(
    const float* __restrict__ w_loc, unsigned short* __restrict__ wlp) {
    const int i = blockIdx.x * 256 + threadIdx.x;      // 64*32 = 2048
    const int lane = i & 31, nt = i >> 5;
    const int d = nt * 16 + (lane & 15);
    const int kb = (lane >> 4) << 3;
    unsigned short* o = wlp + (size_t)i * 16;
    #pragma unroll
    for (int j = 0; j < 16; ++j) o[j] = f2bf(w_loc[d * CC + kb + j + ((j >> 3) << 3)]);
}

// ---------------------------------------------------------------------------
// Conv1d(prev_attn) computed directly into A-fragment (16-bit 16x32) layout:
//   per (b, v-tile, lane): m = lane&15 -> v = vt*16+m ; kb = (lane>=16)*8
//   halves j=0..7 -> c = kb+j ; j=8..15 -> c = kb+16+(j-8)
// ---------------------------------------------------------------------------
__global__ __launch_bounds__(256) void pack_conv_kernel(
    const float* __restrict__ prev, const float* __restrict__ conv_w,
    const float* __restrict__ conv_b, unsigned short* __restrict__ cvp) {
    const int i = blockIdx.x * 256 + threadIdx.x;      // 32*128*32 = 131072
    const int lane = i & 31, vt = (i >> 5) & 127, b = i >> 12;
    const int v = vt * 16 + (lane & 15);
    const int kb = (lane >> 4) << 3;
    const float pm1 = (v > 0)      ? prev[(size_t)b * VL + v - 1] : 0.f;
    const float p00 =               prev[(size_t)b * VL + v];
    const float pp1 = (v < VL - 1) ? prev[(size_t)b * VL + v + 1] : 0.f;
    unsigned short* o = cvp + (size_t)i * 16;
    #pragma unroll
    for (int j = 0; j < 8; ++j) {
        int c0 = kb + j, c1 = kb + 16 + j;
        float f0 = conv_b[c0] + conv_w[c0*3+0]*pm1 + conv_w[c0*3+1]*p00 + conv_w[c0*3+2]*pp1;
        float f1 = conv_b[c1] + conv_w[c1*3+0]*pm1 + conv_w[c1*3+1]*p00 + conv_w[c1*3+2]*pp1;
        o[j]     = f2bf(f0);
        o[8 + j] = f2bf(f1);
    }
}

// ---------------------------------------------------------------------------
// Fused score kernel: per block = (b, 64 value rows).
//   acc = qp + bias  (init)
//   acc += value(bf16) x w_v(bf16)   [32 WMMA K-chunks]
//   acc += conv_feat(bf16) x w_loc(bf16)  [1 WMMA K-chunk]
//   score = sum_d tanh(acc)*w_score[d] ; s = sigmoid(score + b_score)
// Deterministic reductions (shfl + fixed-order LDS sums, no float atomics).
// ---------------------------------------------------------------------------
#define LDSROW 1032   // 1024 bf16 + 8 pad (bank-conflict skew)

__global__ __launch_bounds__(256) void score_kernel(
    const float* __restrict__ value, const float* __restrict__ qpb,
    const unsigned short* __restrict__ wvp, const unsigned short* __restrict__ wlp,
    const unsigned short* __restrict__ cvp, const float* __restrict__ w_score,
    const float* __restrict__ b_score, float* __restrict__ s_ws,
    float* __restrict__ ssum_part) {
    __shared__ unsigned short ldsA[64 * LDSROW];   // 132 KB of the 320 KB WGP LDS
    __shared__ float lds_sc[8 * 64];
    __shared__ float lds_s[64];

    const int tid  = threadIdx.x;
    const int b    = blockIdx.x >> 5;
    const int vblk = blockIdx.x & 31;
    const int v0   = vblk * 64;
    const int wv   = tid >> 5;
    const int lane = tid & 31;

    // Stage 64 value rows into LDS as bf16 (coalesced float4 reads).
    {
        const float* src = value + ((size_t)b * VL + v0) * HD;
        for (int li = tid; li < 64 * 256; li += 256) {
            const int row = li >> 8, c4 = li & 255;
            const float4 f = *(const float4*)(src + (size_t)row * HD + c4 * 4);
            uint2 pk;
            pk.x = (unsigned)f2bf(f.x) | ((unsigned)f2bf(f.y) << 16);
            pk.y = (unsigned)f2bf(f.z) | ((unsigned)f2bf(f.w) << 16);
            *(uint2*)&ldsA[row * LDSROW + c4 * 4] = pk;
        }
    }
    __syncthreads();

    float psum[4][8];
    #pragma unroll
    for (int mt = 0; mt < 4; ++mt)
        #pragma unroll
        for (int r = 0; r < 8; ++r) psum[mt][r] = 0.f;

    const int mrow = lane & 15;
    const int kb_a = (lane >> 4) << 3;

    for (int pr = 0; pr < 4; ++pr) {          // each wave: 4 pairs of n-tiles
        const int nt0 = pr * 8 + wv;
        const int nt1 = nt0 + 32;
        const int d0 = nt0 * 16 + mrow;
        const int d1 = nt1 * 16 + mrow;
        v8f acc0[4], acc1[4];
        {
            const float q0 = qpb[b * DM + d0];
            const float q1 = qpb[b * DM + d1];
            #pragma unroll
            for (int mt = 0; mt < 4; ++mt)
                #pragma unroll
                for (int r = 0; r < 8; ++r) { acc0[mt][r] = q0; acc1[mt][r] = q1; }
        }
        for (int kc = 0; kc < 32; ++kc) {     // K = 1024 over hidden dim
            const v16bf bf0 = *(const v16bf*)(wvp + (((size_t)nt0 * 32 + kc) * 32 + lane) * 16);
            const v16bf bf1 = *(const v16bf*)(wvp + (((size_t)nt1 * 32 + kc) * 32 + lane) * 16);
            #pragma unroll
            for (int mt = 0; mt < 4; ++mt) {
                const unsigned short* ap = &ldsA[(mt * 16 + mrow) * LDSROW + kc * 32 + kb_a];
                FragU af;
                af.u[0] = *(const uint4*)ap;
                af.u[1] = *(const uint4*)(ap + 16);
                acc0[mt] = __builtin_amdgcn_wmma_f32_16x16x32_bf16(false, af.v, false, bf0,
                                                                   (short)0, acc0[mt], false, false);
                acc1[mt] = __builtin_amdgcn_wmma_f32_16x16x32_bf16(false, af.v, false, bf1,
                                                                   (short)0, acc1[mt], false, false);
            }
        }
        {   // location term: one extra K=32 chunk
            const v16bf bl0 = *(const v16bf*)(wlp + ((size_t)nt0 * 32 + lane) * 16);
            const v16bf bl1 = *(const v16bf*)(wlp + ((size_t)nt1 * 32 + lane) * 16);
            #pragma unroll
            for (int mt = 0; mt < 4; ++mt) {
                const v16bf ac = *(const v16bf*)(cvp + (((size_t)b * 128 + (v0 >> 4) + mt) * 32 + lane) * 16);
                acc0[mt] = __builtin_amdgcn_wmma_f32_16x16x32_bf16(false, ac, false, bl0,
                                                                   (short)0, acc0[mt], false, false);
                acc1[mt] = __builtin_amdgcn_wmma_f32_16x16x32_bf16(false, ac, false, bl1,
                                                                   (short)0, acc1[mt], false, false);
            }
        }
        const float ws0 = w_score[d0];
        const float ws1 = w_score[d1];
        #pragma unroll
        for (int mt = 0; mt < 4; ++mt)
            #pragma unroll
            for (int r = 0; r < 8; ++r)
                psum[mt][r] += tanhf(acc0[mt][r]) * ws0 + tanhf(acc1[mt][r]) * ws1;
    }

    // Reduce over the 16 lanes sharing a row; lanes 0/16 publish rows r / r+8.
    #pragma unroll
    for (int mt = 0; mt < 4; ++mt) {
        #pragma unroll
        for (int r = 0; r < 8; ++r) {
            float p = psum[mt][r];
            p += __shfl_xor(p, 8, 32);
            p += __shfl_xor(p, 4, 32);
            p += __shfl_xor(p, 2, 32);
            p += __shfl_xor(p, 1, 32);
            if ((lane & 15) == 0)
                lds_sc[wv * 64 + mt * 16 + r + ((lane >> 4) << 3)] = p;
        }
    }
    __syncthreads();
    if (tid < 64) {
        float sc = 0.f;
        #pragma unroll
        for (int w2 = 0; w2 < 8; ++w2) sc += lds_sc[w2 * 64 + tid];
        sc += b_score[0];
        const float s = 1.0f / (1.0f + expf(-sc));
        s_ws[(size_t)b * VL + v0 + tid] = s;
        lds_s[tid] = s;
    }
    __syncthreads();
    if (tid == 0) {
        float t = 0.f;
        for (int i = 0; i < 64; ++i) t += lds_s[i];
        ssum_part[b * 32 + vblk] = t;
    }
}

// ---------------------------------------------------------------------------
__global__ void ssum_kernel(const float* __restrict__ part, float* __restrict__ ssum) {
    const int b = threadIdx.x;
    if (b < BB) {
        float t = 0.f;
        for (int j = 0; j < 32; ++j) t += part[b * 32 + j];
        ssum[b] = t;
    }
}

__global__ __launch_bounds__(256) void attn_kernel(
    const float* __restrict__ s_ws, const float* __restrict__ ssum,
    float* __restrict__ attn) {
    const int i = blockIdx.x * 256 + threadIdx.x;      // < BB*VL
    attn[i] = s_ws[i] / ssum[i >> 11];
}

// context partials: ctx_part[vc][b*HD + h] = sum over 256 v of s*value
__global__ __launch_bounds__(256) void ctx_kernel(
    const float* __restrict__ s_ws, const float* __restrict__ value,
    float* __restrict__ ctx_part) {
    const int bx = blockIdx.x;                 // 1024
    const int b = bx >> 5, hc = (bx >> 3) & 3, vc = bx & 7;
    const int h = hc * 256 + threadIdx.x;
    const float* vb = value + ((size_t)b * VL + vc * 256) * HD + h;
    const float* sb = s_ws + (size_t)b * VL + vc * 256;
    float acc = 0.f;
    #pragma unroll 4
    for (int v = 0; v < 256; ++v) acc += sb[v] * vb[(size_t)v * HD];
    ctx_part[((size_t)vc * BB + b) * HD + h] = acc;
}

// output[b,h] = b_out[h] + sum_k [ctx/ssum , query] (2H) * w_out[h,k]
__global__ __launch_bounds__(256) void out_kernel(
    const float* __restrict__ ctx_part, const float* __restrict__ ssum,
    const float* __restrict__ query, const float* __restrict__ w_out,
    const float* __restrict__ b_out, float* __restrict__ out) {
    __shared__ float comb[2 * HD];
    const int b = blockIdx.x, tid = threadIdx.x;
    const float inv = 1.0f / ssum[b];
    for (int i = tid; i < HD; i += 256) {
        float c = 0.f;
        #pragma unroll
        for (int vc = 0; vc < 8; ++vc) c += ctx_part[((size_t)vc * BB + b) * HD + i];
        comb[i] = c * inv;
        comb[HD + i] = query[b * HD + i];
    }
    __syncthreads();
    const int wv = tid >> 5, lane = tid & 31;
    for (int h = wv; h < HD; h += 8) {
        const float* wr = w_out + (size_t)h * (2 * HD);
        float acc = 0.f;
        for (int k = lane; k < 2 * HD; k += 32) acc += comb[k] * wr[k];
        acc += __shfl_xor(acc, 16, 32);
        acc += __shfl_xor(acc, 8, 32);
        acc += __shfl_xor(acc, 4, 32);
        acc += __shfl_xor(acc, 2, 32);
        acc += __shfl_xor(acc, 1, 32);
        if (lane == 0) out[(size_t)b * HD + h] = acc + b_out[h];
    }
}

// ---------------------------------------------------------------------------
extern "C" void kernel_launch(void* const* d_in, const int* in_sizes, int n_in,
                              void* d_out, int out_size, void* d_ws, size_t ws_size,
                              hipStream_t stream) {
    const float* query   = (const float*)d_in[0];
    const float* value   = (const float*)d_in[1];
    const float* prev    = (const float*)d_in[2];
    const float* conv_w  = (const float*)d_in[3];
    const float* conv_b  = (const float*)d_in[4];
    const float* w_loc   = (const float*)d_in[5];
    const float* w_q     = (const float*)d_in[6];
    const float* w_v     = (const float*)d_in[7];
    const float* bias    = (const float*)d_in[8];
    const float* w_score = (const float*)d_in[9];
    const float* b_score = (const float*)d_in[10];
    const float* w_out   = (const float*)d_in[11];
    const float* b_out   = (const float*)d_in[12];

    float* out  = (float*)d_out;           // [0, 32768): output ; [32768, 98304): attn
    char*  ws   = (char*)d_ws;

    // workspace layout (256B-aligned offsets)
    float*          qpb       = (float*)(ws + 0);                     // 128 KB
    unsigned short* wvp       = (unsigned short*)(ws + 131072);       // 2 MB
    unsigned short* wlp       = (unsigned short*)(ws + 2228224);      // 64 KB
    unsigned short* cvp       = (unsigned short*)(ws + 2293760);      // 4 MB
    float*          s_ws      = (float*)(ws + 6488064);               // 256 KB
    float*          ctx_part  = (float*)(ws + 6750208);               // 1 MB
    float*          ssum_part = (float*)(ws + 7798784);               // 4 KB
    float*          ssum      = (float*)(ws + 7802880);               // 128 B

    qpb_kernel      <<<BB,  256, 0, stream>>>(query, w_q, bias, qpb);
    pack_wv_kernel  <<<256, 256, 0, stream>>>(w_v, wvp);
    pack_wloc_kernel<<<8,   256, 0, stream>>>(w_loc, wlp);
    pack_conv_kernel<<<512, 256, 0, stream>>>(prev, conv_w, conv_b, cvp);
    score_kernel    <<<BB * 32, 256, 0, stream>>>(value, qpb, wvp, wlp, cvp,
                                                  w_score, b_score, s_ws, ssum_part);
    ssum_kernel     <<<1,   32,  0, stream>>>(ssum_part, ssum);
    attn_kernel     <<<(BB * VL) / 256, 256, 0, stream>>>(s_ws, ssum, out + BB * HD);
    ctx_kernel      <<<1024, 256, 0, stream>>>(s_ws, value, ctx_part);
    out_kernel      <<<BB,  256, 0, stream>>>(ctx_part, ssum, query, w_out, b_out, out);
}
